// Serialized_Attention_19971597926953
// MI455X (gfx1250) — matrix operand order repaired and checked by reading, not compiled, not run on hardware
//
#include <hip/hip_runtime.h>

// ---------------- problem constants ----------------
#define BATCH     8
#define NTOK      4096
#define NTOK_PAD  4120        // 4096 + 24 pad
#define DIM       1024
#define QKV_DIM   3072
#define NHEAD     16
#define HDIM      64
#define NPATCH    206         // 103 blocks * 2 groups
#define PSIZE     20          // tokens per patch
// GEMM tiling
#define TM        128
#define TN        128
#define TK        32
#define LSTR      40          // LDS row stride in bf16 elements (pad vs bank conflicts)

typedef __attribute__((ext_vector_type(16))) __bf16        v16bf;
typedef __attribute__((ext_vector_type(8)))  float         v8f;
typedef __attribute__((ext_vector_type(4)))  float         v4f;
typedef __attribute__((ext_vector_type(4)))  unsigned int  v4u;
typedef __attribute__((ext_vector_type(2)))  unsigned int  v2u;
typedef __attribute__((ext_vector_type(4)))  int           v4i;

union Frag16 { v16bf v; v4u q[2]; unsigned short u[16]; };

__device__ __forceinline__ unsigned short f2bf(float f) {
    unsigned int u = __float_as_uint(f);
    unsigned int r = u + 0x7FFFu + ((u >> 16) & 1u);   // round-to-nearest-even
    return (unsigned short)(r >> 16);
}

// pack 4 f32 -> 4 bf16 in 2 dwords
__device__ __forceinline__ v2u pack4(v4f f) {
    v2u r;
    r.x = (unsigned int)f2bf(f.x) | ((unsigned int)f2bf(f.y) << 16);
    r.y = (unsigned int)f2bf(f.z) | ((unsigned int)f2bf(f.w) << 16);
    return r;
}

// ---- gfx1250 async global->LDS path (probe via __has_builtin) ----
#if defined(__has_builtin)
#if __has_builtin(__builtin_amdgcn_global_load_async_to_lds_b128) && \
    __has_builtin(__builtin_amdgcn_s_wait_asynccnt)
#define HAVE_ASYNC_LDS 1
#endif
#endif
#ifndef HAVE_ASYNC_LDS
#define HAVE_ASYNC_LDS 0
#endif

#if HAVE_ASYNC_LDS
// param 1: int4 in addrspace(1) (prints as "int4 __device__*")
// param 2: int4 in addrspace(3) (LDS)
typedef __attribute__((address_space(1))) v4i g_v4i;
typedef __attribute__((address_space(3))) v4i s_v4i;
#endif

// =====================================================================
// Kernel 1: QKV projection.  C[m,n] = sum_k A[m,k] * Wqkv[n,k]
//   A = x (f32, padded rows are zero), out = bf16 qkv [B][4120][3072]
// =====================================================================
__global__ __launch_bounds__(256) void qkv_gemm_kernel(
    const float* __restrict__ x, const float* __restrict__ Wqkv,
    unsigned short* __restrict__ qkv)
{
    __shared__ unsigned short Asm[TM * LSTR];
    __shared__ unsigned short Bsm[TN * LSTR];

    const int Mtot = BATCH * NTOK_PAD;
    const int mt = blockIdx.x, nt = blockIdx.y;
    const int tid = threadIdx.x, lane = tid & 31, wave = tid >> 5;
    const int mwave = (wave & 3) * 32;       // wave's M offset inside tile
    const int nwave = (wave >> 2) * 64;      // wave's N offset inside tile
    const int nlo = lane & 15, sel = lane >> 4;

    v8f acc[2][4];
    #pragma unroll
    for (int i = 0; i < 2; ++i)
        #pragma unroll
        for (int j = 0; j < 4; ++j) acc[i][j] = (v8f){};

    // cooperative-load coordinates: 2 threads per row, 16 cols each
    const int ar = tid >> 1, ac = (tid & 1) * 16;
    const int grow = mt * TM + ar;
    bool avalid = false; const float* asrc = nullptr;
    if (grow < Mtot) {
        int b = grow / NTOK_PAD, t = grow - b * NTOK_PAD;
        if (t < NTOK) { avalid = true; asrc = x + ((size_t)(b * NTOK + t)) * DIM + ac; }
    }
    const float* bsrc = Wqkv + (size_t)(nt * TN + ar) * DIM + ac;
    const float* pfa  = avalid ? asrc : x;   // always-safe prefetch address

    for (int k0 = 0; k0 < DIM; k0 += TK) {
        if (k0 + TK < DIM) {
            __builtin_prefetch(pfa  + k0 + TK, 0, 3);
            __builtin_prefetch(bsrc + k0 + TK, 0, 3);
        }
        #pragma unroll
        for (int j = 0; j < 4; ++j) {
            v4f fa = (v4f){0.f, 0.f, 0.f, 0.f};
            if (avalid) fa = *(const v4f*)(asrc + k0 + 4 * j);
            v4f fb = *(const v4f*)(bsrc + k0 + 4 * j);
            *(v2u*)&Asm[ar * LSTR + ac + 4 * j] = pack4(fa);
            *(v2u*)&Bsm[ar * LSTR + ac + 4 * j] = pack4(fb);
        }
        __syncthreads();

        Frag16 af[2], bf[4];
        const int kh = sel * 8, kb = sel * 16;
        #pragma unroll
        for (int i = 0; i < 2; ++i) {
            int m = mwave + i * 16 + nlo;
            af[i].q[0] = *(const v4u*)&Asm[m * LSTR + kh];
            af[i].q[1] = *(const v4u*)&Asm[m * LSTR + 16 + kh];
        }
        #pragma unroll
        for (int j = 0; j < 4; ++j) {
            int n = nwave + j * 16 + nlo;
            bf[j].q[0] = *(const v4u*)&Bsm[n * LSTR + kh];
            bf[j].q[1] = *(const v4u*)&Bsm[n * LSTR + 16 + kh];
        }
        #pragma unroll
        for (int i = 0; i < 2; ++i)
            #pragma unroll
            for (int j = 0; j < 4; ++j)
                acc[i][j] = __builtin_amdgcn_wmma_f32_16x16x32_bf16(
                    false, af[i].v, false, bf[j].v, (short)0, acc[i][j], false, false);
        __syncthreads();
    }

    // store bf16 qkv
    #pragma unroll
    for (int i = 0; i < 2; ++i)
        #pragma unroll
        for (int j = 0; j < 4; ++j)
            #pragma unroll
            for (int e = 0; e < 8; ++e) {
                int row = mt * TM + mwave + i * 16 + sel * 8 + e;
                if (row < Mtot) {
                    int col = nt * TN + nwave + j * 16 + nlo;
                    qkv[(size_t)row * QKV_DIM + col] = f2bf(acc[i][j][e]);
                }
            }
}

// =====================================================================
// Kernel 2: per-patch attention.  1 WG per (b, patch r), 1 wave per head.
//   patch r = 2*blk + g -> token(s) = 40*blk + g + 2*s,  s in [0,20)
//   Pad S and T to 32, K(head)=64.  All GEMMs via bf16 WMMA, f32 softmax.
// =====================================================================
__global__ __launch_bounds__(512) void attn_kernel(
    const unsigned short* __restrict__ qkv, unsigned short* __restrict__ aout)
{
    __shared__ unsigned short Psm[NHEAD * 32 * LSTR];   // per-wave 32x32 P tile

    const int br = blockIdx.x;
    const int b = br / NPATCH, r = br - b * NPATCH;
    const int blk = r >> 1, g = r & 1;
    const int h = threadIdx.x >> 5;       // wave index == head
    const int lane = threadIdx.x & 31;
    const int nlo = lane & 15, sel = lane >> 4;
    const size_t base = (size_t)b * NTOK_PAD * QKV_DIM;
    unsigned short* P = &Psm[h * 32 * LSTR];

    const v4u z4 = {0u, 0u, 0u, 0u};

    // ---- load Q as A-fragments (rows=s, K=head dim, 2 k-steps) ----
    Frag16 qa[2][2], kf[2][2];
    #pragma unroll
    for (int si = 0; si < 2; ++si) {
        int s = si * 16 + nlo;
        bool vld = s < PSIZE;
        int tq = 40 * blk + g + 2 * s;
        const unsigned short* qrow = qkv + base + (size_t)tq * QKV_DIM + h * HDIM;
        const unsigned short* krow = qkv + base + (size_t)tq * QKV_DIM + DIM + h * HDIM;
        #pragma unroll
        for (int kk = 0; kk < 2; ++kk) {
            int kh = sel * 8, kb = sel * 16;
            if (vld) {
                qa[si][kk].q[0] = *(const v4u*)(qrow + kk * 32 + kh);
                qa[si][kk].q[1] = *(const v4u*)(qrow + kk * 32 + 16 + kh);
                kf[si][kk].q[0] = *(const v4u*)(krow + kk * 32 + kh);      // B-frag: n=t tile si
                kf[si][kk].q[1] = *(const v4u*)(krow + kk * 32 + 16 + kh);
            } else {
                qa[si][kk].q[0] = z4; qa[si][kk].q[1] = z4;
                kf[si][kk].q[0] = z4; kf[si][kk].q[1] = z4;
            }
        }
    }

    // ---- scores S = Q K^T  (32x32 padded) ----
    v8f sc[2][2];
    #pragma unroll
    for (int si = 0; si < 2; ++si)
        #pragma unroll
        for (int ti = 0; ti < 2; ++ti) {
            sc[si][ti] = (v8f){};
            #pragma unroll
            for (int kk = 0; kk < 2; ++kk)
                sc[si][ti] = __builtin_amdgcn_wmma_f32_16x16x32_bf16(
                    false, qa[si][kk].v, false, kf[ti][kk].v, (short)0, sc[si][ti], false, false);
        }

    // ---- scale, mask, softmax (rows split across 16-lane halves) ----
    const float scale = 0.125f;   // 64^-0.5
    #pragma unroll
    for (int si = 0; si < 2; ++si) {
        #pragma unroll
        for (int e = 0; e < 8; ++e) {
            float v0 = sc[si][0][e] * scale;
            float v1 = sc[si][1][e] * scale;
            if (nlo >= PSIZE - 16) v1 = -1e30f;      // cols 16+nlo >= 20 masked
            float m = fmaxf(v0, v1);
            #pragma unroll
            for (int off = 8; off >= 1; off >>= 1) m = fmaxf(m, __shfl_xor(m, off, 32));
            float p0 = __expf(v0 - m), p1 = __expf(v1 - m);
            float sum = p0 + p1;
            #pragma unroll
            for (int off = 8; off >= 1; off >>= 1) sum += __shfl_xor(sum, off, 32);
            float inv = __frcp_rn(sum);
            int row = si * 16 + sel * 8 + e;
            P[row * LSTR + nlo]      = f2bf(p0 * inv);
            P[row * LSTR + 16 + nlo] = f2bf(p1 * inv);
        }
    }
    // same-wave LDS store->load is kept in order by hardware; no barrier needed

    // ---- P as A-fragments (K = t, one 32-step) ----
    Frag16 pa[2];
    #pragma unroll
    for (int si = 0; si < 2; ++si) {
        int row = si * 16 + nlo, kh = sel * 8;
        pa[si].q[0] = *(const v4u*)&P[row * LSTR + kh];
        pa[si].q[1] = *(const v4u*)&P[row * LSTR + 16 + kh];
    }

    // ---- V as B-fragments (K = t, N = head dim, 4 n-tiles) ----
    Frag16 vf[4];
    const int kh = sel * 8;
    #pragma unroll
    for (int nn = 0; nn < 4; ++nn) {
        int d = nn * 16 + nlo;
        #pragma unroll
        for (int j = 0; j < 16; ++j) {
            int t_ = kh + j + (j >= 8 ? 8 : 0);
            vf[nn].u[j] = (t_ < PSIZE)
                ? qkv[base + (size_t)(40 * blk + g + 2 * t_) * QKV_DIM + 2 * DIM + h * HDIM + d]
                : (unsigned short)0;
        }
    }

    // ---- O = P V  (32x64) ----
    v8f oa[2][4];
    #pragma unroll
    for (int si = 0; si < 2; ++si)
        #pragma unroll
        for (int nn = 0; nn < 4; ++nn) {
            oa[si][nn] = (v8f){};
            oa[si][nn] = __builtin_amdgcn_wmma_f32_16x16x32_bf16(
                false, pa[si].v, false, vf[nn].v, (short)0, oa[si][nn], false, false);
        }

    // ---- scatter O back to token order (bf16) ----
    #pragma unroll
    for (int si = 0; si < 2; ++si)
        #pragma unroll
        for (int nn = 0; nn < 4; ++nn)
            #pragma unroll
            for (int e = 0; e < 8; ++e) {
                int s = si * 16 + sel * 8 + e;
                if (s < PSIZE) {
                    int t = 40 * blk + g + 2 * s;
                    aout[((size_t)b * NTOK_PAD + t) * DIM + h * HDIM + nn * 16 + nlo] =
                        f2bf(oa[si][nn][e]);
                }
            }
}

// =====================================================================
// Kernel 3: output projection.  out[m,n] = sum_k A[m,k]*Wout[n,k] + bout[n]
//   A = bf16 attention output (skip pad tokens), out f32.
//   A-tile staging uses GLOBAL_LOAD_ASYNC_TO_LDS_B128 when available.
// =====================================================================
__global__ __launch_bounds__(256) void out_gemm_kernel(
    const unsigned short* __restrict__ aout, const float* __restrict__ Wout,
    const float* __restrict__ bout, float* __restrict__ out)
{
    __shared__ unsigned short Asm[TM * LSTR];
    __shared__ unsigned short Bsm[TN * LSTR];

    const int mt = blockIdx.x, nt = blockIdx.y;
    const int tid = threadIdx.x, lane = tid & 31, wave = tid >> 5;
    const int mwave = (wave & 3) * 32, nwave = (wave >> 2) * 64;
    const int nlo = lane & 15, sel = lane >> 4;

    v8f acc[2][4];
    #pragma unroll
    for (int i = 0; i < 2; ++i)
        #pragma unroll
        for (int j = 0; j < 4; ++j) acc[i][j] = (v8f){};

    const int ar = tid >> 1, ac = (tid & 1) * 16;
    const int grow = mt * TM + ar;                 // < 32768 always
    const int b = grow >> 12, t = grow & 4095;
    const unsigned short* asrc = aout + ((size_t)(b * NTOK_PAD + t)) * DIM + ac;
    const float* bsrc = Wout + (size_t)(nt * TN + ar) * DIM + ac;

    for (int k0 = 0; k0 < DIM; k0 += TK) {
        if (k0 + TK < DIM) {
            __builtin_prefetch(asrc + k0 + TK, 0, 3);
            __builtin_prefetch(bsrc + k0 + TK, 0, 3);
        }
#if HAVE_ASYNC_LDS
        // async DMA copy of the bf16 A-tile straight into LDS (ASYNCcnt path)
        __builtin_amdgcn_global_load_async_to_lds_b128(
            (g_v4i*)(asrc + k0),     (s_v4i*)&Asm[ar * LSTR + ac],     0, 0);
        __builtin_amdgcn_global_load_async_to_lds_b128(
            (g_v4i*)(asrc + k0 + 8), (s_v4i*)&Asm[ar * LSTR + ac + 8], 0, 0);
#else
        *(v4u*)&Asm[ar * LSTR + ac]     = *(const v4u*)(asrc + k0);
        *(v4u*)&Asm[ar * LSTR + ac + 8] = *(const v4u*)(asrc + k0 + 8);
#endif
        #pragma unroll
        for (int j = 0; j < 4; ++j) {
            v4f fb = *(const v4f*)(bsrc + k0 + 4 * j);
            *(v2u*)&Bsm[ar * LSTR + ac + 4 * j] = pack4(fb);
        }
#if HAVE_ASYNC_LDS
        __builtin_amdgcn_s_wait_asynccnt(0);
#endif
        __syncthreads();

        Frag16 af[2], bf[4];
        const int kh = sel * 8, kb = sel * 16;
        #pragma unroll
        for (int i = 0; i < 2; ++i) {
            int m = mwave + i * 16 + nlo;
            af[i].q[0] = *(const v4u*)&Asm[m * LSTR + kh];
            af[i].q[1] = *(const v4u*)&Asm[m * LSTR + 16 + kh];
        }
        #pragma unroll
        for (int j = 0; j < 4; ++j) {
            int n = nwave + j * 16 + nlo;
            bf[j].q[0] = *(const v4u*)&Bsm[n * LSTR + kh];
            bf[j].q[1] = *(const v4u*)&Bsm[n * LSTR + 16 + kh];
        }
        #pragma unroll
        for (int i = 0; i < 2; ++i)
            #pragma unroll
            for (int j = 0; j < 4; ++j)
                acc[i][j] = __builtin_amdgcn_wmma_f32_16x16x32_bf16(
                    false, af[i].v, false, bf[j].v, (short)0, acc[i][j], false, false);
        __syncthreads();
    }

    #pragma unroll
    for (int j = 0; j < 4; ++j) {
        int col = nt * TN + nwave + j * 16 + nlo;
        float bias = bout[col];
        #pragma unroll
        for (int i = 0; i < 2; ++i)
            #pragma unroll
            for (int e = 0; e < 8; ++e) {
                int row = mt * TM + mwave + i * 16 + sel * 8 + e;
                out[(size_t)row * DIM + col] = acc[i][j][e] + bias;
            }
    }
}

// =====================================================================
extern "C" void kernel_launch(void* const* d_in, const int* in_sizes, int n_in,
                              void* d_out, int out_size, void* d_ws, size_t ws_size,
                              hipStream_t stream) {
    const float* x    = (const float*)d_in[0];   // [8,4096,1024]
    const float* Wqkv = (const float*)d_in[1];   // [3072,1024]
    const float* Wout = (const float*)d_in[2];   // [1024,1024]
    const float* bout = (const float*)d_in[3];   // [1024]
    float* out = (float*)d_out;                  // [8,4096,1024]

    const size_t qkv_bytes = (size_t)BATCH * NTOK_PAD * QKV_DIM * sizeof(unsigned short);
    unsigned short* qkv_bf  = (unsigned short*)d_ws;
    unsigned short* attn_bf = (unsigned short*)((char*)d_ws + qkv_bytes);

    // 1) QKV projection: M=8*4120 (258 tiles), N=3072 (24 tiles)
    qkv_gemm_kernel<<<dim3((BATCH * NTOK_PAD + TM - 1) / TM, QKV_DIM / TN), 256, 0, stream>>>(
        x, Wqkv, qkv_bf);

    // 2) per-patch attention: 1 WG per (b, patch), 16 waves = 16 heads
    attn_kernel<<<dim3(BATCH * NPATCH), 512, 0, stream>>>(qkv_bf, attn_bf);

    // 3) output projection: M=8*4096 (256 tiles), N=1024 (8 tiles)
    out_gemm_kernel<<<dim3((BATCH * NTOK) / TM, DIM / TN), 256, 0, stream>>>(
        attn_bf, Wout, bout, out);
}
